// LSTMNet_7464653160926
// MI455X (gfx1250) — matrix operand order repaired and checked by reading, not compiled, not run on hardware
//
#include <hip/hip_runtime.h>

// ---------------------------------------------------------------------------
// LSTMNet for MI455X (gfx1250): bf16 WMMA 16x16x32 for all matmuls.
// Batch tile of 16 rows per block; 6 waves, one per (lstm, direction).
// A-fragments for x and h live in LDS in per-lane fragment order (2x
// ds_load_b128 per operand). Gate math uses contiguous-j lane ownership so
// gate reads are b128 loads and h write-back is one b128 + one b64 store.
// ---------------------------------------------------------------------------

typedef __attribute__((ext_vector_type(16))) __bf16 v16bf;
typedef __attribute__((ext_vector_type(8)))  float  v8f;

#define H_    24
#define IN_   20
#define L_    69
#define D_    64
#define SIXH  144
#define TCOMP 41   // compacted used timesteps: 9 (pep) + 15 (cdra) + 17 (cdrb)

// LDS layout (bytes)
#define OFF_XSF   0                       // ushort xsf[41][32][16]    = 41984
#define OFF_GATES 41984                   // float  gates[6][16][96]   = 36864
#define OFF_HA    (41984 + 36864)         // ushort hA[6][32][16]      =  6144
#define OFF_HCAT  (OFF_HA + 6144)         // float  hcat[16][144]      =  9216
#define OFF_H1    (OFF_HCAT + 9216)       // float  h1[16][64]         =  4096
#define SMEM_BYTES (OFF_H1 + 4096)        // 98304

static __device__ __forceinline__ unsigned short f2bf_bits(float f) {
    unsigned int u = __float_as_uint(f);
    unsigned int r = (u + 0x7FFFu + ((u >> 16) & 1u)) >> 16;  // RNE
    return (unsigned short)r;
}
static __device__ __forceinline__ float bf_bits2f(unsigned short s) {
    return __uint_as_float(((unsigned int)s) << 16);
}
static __device__ __forceinline__ __bf16 bits2bf(unsigned short s) {
    union { unsigned short u; __bf16 b; } c; c.u = s; return c.b;
}
static __device__ __forceinline__ __bf16 f2bf(float f) { return bits2bf(f2bf_bits(f)); }

static __device__ __forceinline__ float fsig(float x)  { return 1.0f / (1.0f + __expf(-x)); }
static __device__ __forceinline__ float ftanh_(float x){ return 1.0f - 2.0f / (__expf(2.0f * x) + 1.0f); }

// A-fragment (16-bit, 16x32) K index for element e, lane-half h (ISA 7.12.2)
static __device__ __forceinline__ int amap(int e, int h) {
    return (e < 8) ? (h * 8 + e) : (h * 8 + e + 8);
}

// Load a 32B per-lane A/B fragment (16 bf16) from LDS with two b128 loads.
static __device__ __forceinline__ v16bf load_frag16(const unsigned short* p) {
    union { v16bf v; uint4 q[2]; } u;
    u.q[0] = ((const uint4*)p)[0];
    u.q[1] = ((const uint4*)p)[1];
    return u.v;
}

// Load 12 contiguous floats (16B-aligned) with three b128 loads.
struct f12 { float f[12]; };
static __device__ __forceinline__ f12 load_f12(const float* p) {
    union { f12 r; float4 q[3]; } u;
    u.q[0] = ((const float4*)p)[0];
    u.q[1] = ((const float4*)p)[1];
    u.q[2] = ((const float4*)p)[2];
    return u.r;
}

__global__ __launch_bounds__(192)
void lstmnet_cdna5_kernel(
    const float* __restrict__ x,
    const float* __restrict__ Wih0, const float* __restrict__ Whh0,
    const float* __restrict__ bih0, const float* __restrict__ bhh0,
    const float* __restrict__ Wih1, const float* __restrict__ Whh1,
    const float* __restrict__ bih1, const float* __restrict__ bhh1,
    const float* __restrict__ Wih2, const float* __restrict__ Whh2,
    const float* __restrict__ bih2, const float* __restrict__ bhh2,
    const float* __restrict__ W1,  const float* __restrict__ b1,
    const float* __restrict__ W2,  const float* __restrict__ b2,
    float* __restrict__ out)
{
    extern __shared__ char smem[];
    unsigned short* xsf   = (unsigned short*)(smem + OFF_XSF);
    float*          gates = (float*)         (smem + OFF_GATES);
    unsigned short* hA    = (unsigned short*)(smem + OFF_HA);
    float*          hcat  = (float*)         (smem + OFF_HCAT);
    float*          h1s   = (float*)         (smem + OFF_H1);

    const int tid  = threadIdx.x;
    const int wid  = tid >> 5;          // 0..5
    const int ln   = tid & 31;
    const int half = ln >> 4;
    const int mrow = ln & 15;
    const int ncol = ln & 15;
    const int b0   = blockIdx.x * 16;

    // Warm L2 with the (block-shared) MLP weights while we stage x.
    __builtin_prefetch(W1 + ((tid * 16) % (D_ * SIXH)), 0, 1);

    // ---- Pass 1: zero-fill fragment-order x buffer (covers K padding) ----
    {
        unsigned int* p32 = (unsigned int*)xsf;
        for (int i = tid; i < (TCOMP * 32 * 16) / 2; i += 192) p32[i] = 0u;
    }
    __syncthreads();

    // ---- Pass 2: stage x tile into A-fragment per-lane order -------------
    for (int idx = tid; idx < 16 * IN_ * TCOMP; idx += 192) {
        int s   = idx / (IN_ * TCOMP);
        int rem = idx - s * (IN_ * TCOMP);
        int i   = rem / TCOMP;          // k
        int t   = rem - i * TCOMP;
        int l   = (t < 9) ? t : ((t < 24) ? t + 15 : t + 28);  // pep|cdra|cdrb
        float v = x[((size_t)(b0 + s) * IN_ + i) * L_ + l];
        int lane, e;
        if (i < 8)       { lane = s;      e = i;      }
        else if (i < 16) { lane = s + 16; e = i - 8;  }
        else             { lane = s;      e = i - 8;  }   // i in 16..19
        xsf[(t * 32 + lane) * 16 + e] = f2bf_bits(v);
    }

    // ---- Per-wave LSTM selection ----
    const int lstm = wid >> 1;          // 0=pep 1=cdra 2=cdrb
    const int dirr = wid & 1;           // 0=fwd 1=bwd
    const int T    = (lstm == 0) ? 9  : ((lstm == 1) ? 15 : 17);
    const int segb = (lstm == 0) ? 0  : ((lstm == 1) ? 9  : 24);

    const float* Wih = ((lstm == 0) ? Wih0 : (lstm == 1) ? Wih1 : Wih2) + dirr * 96 * IN_;
    const float* Whh = ((lstm == 0) ? Whh0 : (lstm == 1) ? Whh1 : Whh2) + dirr * 96 * H_;
    const float* bih = ((lstm == 0) ? bih0 : (lstm == 1) ? bih1 : bih2) + dirr * 96;
    const float* bhh = ((lstm == 0) ? bhh0 : (lstm == 1) ? bhh1 : bhh2) + dirr * 96;

    // ---- Weights as WMMA B-fragments held in VGPRs (K padded to 32) ----
    v16bf bWih[6], bWhh[6];
    float biasv[6];
#pragma unroll
    for (int j = 0; j < 6; ++j) {
        const int n = j * 16 + ncol;
#pragma unroll
        for (int e = 0; e < 16; ++e) {
            const int k = amap(e, half);                 // B-frag K map
            bWih[j][e] = f2bf((k < IN_) ? Wih[n * IN_ + k] : 0.0f);
            bWhh[j][e] = f2bf((k < H_)  ? Whh[n * H_  + k] : 0.0f);
        }
        biasv[j] = bih[n] + bhh[n];
    }

    // ---- Init h fragment buffer (zeros incl. K 24..31 padding) and c ----
    unsigned short* hAw = hA + wid * 32 * 16;
    {
        unsigned int* p32 = (unsigned int*)hAw;
        for (int e = ln; e < 256; e += 32) p32[e] = 0u;
    }
    float creg[12];
#pragma unroll
    for (int q = 0; q < 12; ++q) creg[q] = 0.0f;

    float* gw = gates + wid * 16 * 96;
    const unsigned short* hfragp = hAw + ln * 16;

    // Gate-math lane ownership: m = mrow, j in [j0, j0+12), j0 = half*12
    const int j0 = half * 12;
    const float* gr = gw + mrow * 96 + j0;
    // h write-back targets (inverse A-fragment map, contiguous runs):
    //  half==0: j0..7  -> row mrow    elems 0..7  (16B);  j8..11  -> row mrow+16 elems 0..3 (8B)
    //  half==1: j16..23-> row mrow    elems 8..15 (16B);  j12..15 -> row mrow+16 elems 4..7 (8B)
    unsigned short* dst128 = hAw + mrow * 16 + (half ? 8 : 0);
    unsigned short* dst64  = hAw + (mrow + 16) * 16 + (half ? 4 : 0);
    const int qBase128 = half ? 4 : 0;   // creg index feeding the 16B store
    const int qBase64  = half ? 0 : 8;   // creg index feeding the 8B store

    __syncthreads();   // xsf ready

    // ---- Recurrent scan ----
    for (int tt = 0; tt < T; ++tt) {
        const int tc = segb + ((dirr == 0) ? tt : (T - 1 - tt));

        const v16bf ax = load_frag16(xsf + (tc * 32 + ln) * 16);
        const v16bf ah = load_frag16(hfragp);

        v8f acc[6];
#pragma unroll
        for (int j = 0; j < 6; ++j) {
            v8f c;
#pragma unroll
            for (int r = 0; r < 8; ++r) c[r] = biasv[j];
            c = __builtin_amdgcn_wmma_f32_16x16x32_bf16(false, ax, false, bWih[j],
                                                        (short)0, c, false, false);
            c = __builtin_amdgcn_wmma_f32_16x16x32_bf16(false, ah, false, bWhh[j],
                                                        (short)0, c, false, false);
            acc[j] = c;
        }

        // Spill gate pre-activations (C layout: m = r + 8*half, n = j*16 + ncol)
#pragma unroll
        for (int j = 0; j < 6; ++j)
#pragma unroll
            for (int r = 0; r < 8; ++r)
                gw[(r + half * 8) * 96 + j * 16 + ncol] = acc[j][r];

#if __has_builtin(__builtin_amdgcn_s_wait_dscnt)
        __builtin_amdgcn_s_wait_dscnt(0);
#endif

        // Gate readback: 4 gate types x 12 contiguous floats (3x b128 each)
        const f12 iv = load_f12(gr);
        const f12 fv = load_f12(gr + 24);
        const f12 gv = load_f12(gr + 48);
        const f12 ov = load_f12(gr + 72);

        unsigned short hb[12];
#pragma unroll
        for (int q = 0; q < 12; ++q) {
            const float cc = fsig(fv.f[q]) * creg[q] + fsig(iv.f[q]) * ftanh_(gv.f[q]);
            creg[q] = cc;
            hb[q] = f2bf_bits(fsig(ov.f[q]) * ftanh_(cc));
        }

        // Pack and store new h into the A-fragment buffer: 16B + 8B
        {
            uint4 w4;
            w4.x = (unsigned int)hb[qBase128 + 0] | ((unsigned int)hb[qBase128 + 1] << 16);
            w4.y = (unsigned int)hb[qBase128 + 2] | ((unsigned int)hb[qBase128 + 3] << 16);
            w4.z = (unsigned int)hb[qBase128 + 4] | ((unsigned int)hb[qBase128 + 5] << 16);
            w4.w = (unsigned int)hb[qBase128 + 6] | ((unsigned int)hb[qBase128 + 7] << 16);
            *((uint4*)dst128) = w4;
            uint2 w2;
            w2.x = (unsigned int)hb[qBase64 + 0] | ((unsigned int)hb[qBase64 + 1] << 16);
            w2.y = (unsigned int)hb[qBase64 + 2] | ((unsigned int)hb[qBase64 + 3] << 16);
            *((uint2*)dst64) = w2;
        }
    }

    // ---- Final hidden state -> hcat (read back from fragment buffer) ----
#pragma unroll
    for (int e = 0; e < 16; ++e) {
        const int k = amap(e, half);
        if (k < H_)
            hcat[mrow * SIXH + lstm * 48 + dirr * 24 + k] = bf_bits2f(hfragp[e]);
    }

    __syncthreads();   // hcat complete

    // ---- MLP layer 1: relu(hcat(16x144) @ W1^T(144x64) + b1), WMMA bf16 ----
    if (wid < 4) {
        const int nt = wid;
        const int n  = nt * 16 + ncol;
        const float bb = b1[n];
        v8f acc;
#pragma unroll
        for (int r = 0; r < 8; ++r) acc[r] = bb;
#pragma unroll
        for (int kc = 0; kc < 5; ++kc) {
            v16bf aw, bw;
#pragma unroll
            for (int e = 0; e < 16; ++e) {
                const int ka = kc * 32 + amap(e, half);
                const int kb = kc * 32 + amap(e, half);
                aw[e] = f2bf((ka < SIXH) ? hcat[mrow * SIXH + ka] : 0.0f);
                bw[e] = f2bf((kb < SIXH) ? W1[n * SIXH + kb] : 0.0f);
            }
            acc = __builtin_amdgcn_wmma_f32_16x16x32_bf16(false, aw, false, bw,
                                                          (short)0, acc, false, false);
        }
#pragma unroll
        for (int r = 0; r < 8; ++r) {
            const float v = acc[r];
            h1s[(r + half * 8) * D_ + n] = (v > 0.0f) ? v : 0.0f;
        }
    }

    __syncthreads();   // h1 complete

    // ---- MLP layer 2: sigmoid(h1 @ W2^T + b2) ----
    if (tid < 16) {
        float acc = b2[0];
#pragma unroll 8
        for (int k = 0; k < D_; ++k)
            acc = fmaf(h1s[tid * D_ + k], W2[k], acc);
        out[b0 + tid] = fsig(acc);
    }
}

extern "C" void kernel_launch(void* const* d_in, const int* in_sizes, int n_in,
                              void* d_out, int out_size, void* d_ws, size_t ws_size,
                              hipStream_t stream) {
    (void)n_in; (void)d_ws; (void)ws_size; (void)out_size;
    const float* x     = (const float*)d_in[0];
    const float* Wih0  = (const float*)d_in[1];
    const float* Whh0  = (const float*)d_in[2];
    const float* bih0  = (const float*)d_in[3];
    const float* bhh0  = (const float*)d_in[4];
    const float* Wih1  = (const float*)d_in[5];
    const float* Whh1  = (const float*)d_in[6];
    const float* bih1  = (const float*)d_in[7];
    const float* bhh1  = (const float*)d_in[8];
    const float* Wih2  = (const float*)d_in[9];
    const float* Whh2  = (const float*)d_in[10];
    const float* bih2  = (const float*)d_in[11];
    const float* bhh2  = (const float*)d_in[12];
    const float* W1    = (const float*)d_in[13];
    const float* b1    = (const float*)d_in[14];
    const float* W2    = (const float*)d_in[15];
    const float* b2    = (const float*)d_in[16];
    float* out = (float*)d_out;

    const int B = in_sizes[0] / (IN_ * L_);     // 32768
    const int nblocks = B / 16;                  // 2048

    (void)hipFuncSetAttribute((const void*)lstmnet_cdna5_kernel,
                              hipFuncAttributeMaxDynamicSharedMemorySize,
                              SMEM_BYTES);

    lstmnet_cdna5_kernel<<<nblocks, 192, SMEM_BYTES, stream>>>(
        x, Wih0, Whh0, bih0, bhh0, Wih1, Whh1, bih1, bhh1,
        Wih2, Whh2, bih2, bhh2, W1, b1, W2, b2, out);
}
